// MoE_39402029973994
// MI455X (gfx1250) — matrix operand order repaired and checked by reading, not compiled, not run on hardware
//
#include <hip/hip_runtime.h>
#include <hip/hip_bf16.h>
#include <stdint.h>

#define N_TOK 2048
#define HDIM  1024
#define NEXP  16
#define FDIM  1408
#define TWOF  2816

typedef __attribute__((ext_vector_type(8)))  float          v8f;
typedef __attribute__((ext_vector_type(8)))  __bf16         v8bf;
typedef __attribute__((ext_vector_type(16))) __bf16         v16bf;
typedef __attribute__((ext_vector_type(8)))  unsigned short v8us;

// Pack two fp32 -> two bf16 (round-half-up) in one dword: 2x v_add + 1x v_perm_b32.
__device__ __forceinline__ unsigned int pk2bf(float lo, float hi) {
  unsigned int a = __float_as_uint(lo) + 0x8000u;
  unsigned int b = __float_as_uint(hi) + 0x8000u;
  return __builtin_amdgcn_perm(b, a, 0x07060302u);
}
__device__ __forceinline__ unsigned short f2bf(float f) {
  unsigned int u = __float_as_uint(f);
  u += 0x7fffu + ((u >> 16) & 1u);
  return (unsigned short)(u >> 16);
}
__device__ __forceinline__ uint4 pk8bf(float4 a, float4 b) {
  return make_uint4(pk2bf(a.x, a.y), pk2bf(a.z, a.w), pk2bf(b.x, b.y), pk2bf(b.z, b.w));
}

// A-fragment (16x32 bf16): lane holds K {c..c+7, c+16..c+23}, c=(lane>>4)*8  -> 2x ds_read_b128
__device__ __forceinline__ v16bf lds_load_a(const unsigned short* p0) {
  v8bf lo = __builtin_bit_cast(v8bf, *(const v8us*)p0);
  v8bf hi = __builtin_bit_cast(v8bf, *(const v8us*)(p0 + 16));
  return __builtin_shufflevector(lo, hi, 0,1,2,3,4,5,6,7,8,9,10,11,12,13,14,15);
}
// B-fragment (32x16 bf16): lane n=l&15 holds K {c..c+7, c+16..c+23}, c=(l>>4)*8 -> 2x ds_read_b128
__device__ __forceinline__ v16bf lds_load_b(const unsigned short* p0) {
  v8bf lo = __builtin_bit_cast(v8bf, *(const v8us*)p0);
  v8bf hi = __builtin_bit_cast(v8bf, *(const v8us*)(p0 + 16));
  return __builtin_shufflevector(lo, hi, 0,1,2,3,4,5,6,7,8,9,10,11,12,13,14,15);
}

#define WMMA_BF16(A, B, C) \
  __builtin_amdgcn_wmma_f32_16x16x32_bf16(false, (A), false, (B), (short)0, (C), false, false)

// ---------------- Kernel 1: gating (logits, top-2, routing lists) ----------------
__global__ __launch_bounds__(256) void moe_gate(const float* __restrict__ x,
                                                const float* __restrict__ gw,
                                                int* __restrict__ cnt,
                                                int* __restrict__ list_slot,
                                                float* __restrict__ slot_w) {
  const int wv = threadIdx.x >> 5, lane = threadIdx.x & 31;
  const int n = blockIdx.x * 8 + wv;
  const float* xr = x + (size_t)n * HDIM;
  float xv[32];
#pragma unroll
  for (int i = 0; i < 32; ++i) xv[i] = xr[lane + 32 * i];
  float logits[16];
#pragma unroll
  for (int e = 0; e < 16; ++e) {
    const float* gr = gw + (size_t)e * HDIM;
    float s = 0.f;
#pragma unroll
    for (int i = 0; i < 32; ++i) s += xv[i] * gr[lane + 32 * i];
#pragma unroll
    for (int off = 16; off >= 1; off >>= 1) s += __shfl_xor(s, off, 32);
    logits[e] = s;
  }
  int i1 = 0; float m1 = logits[0];
#pragma unroll
  for (int e = 1; e < 16; ++e) if (logits[e] > m1) { m1 = logits[e]; i1 = e; }
  int i2 = (i1 == 0) ? 1 : 0; float m2 = logits[i2];
#pragma unroll
  for (int e = 0; e < 16; ++e) if (e != i1 && logits[e] > m2) { m2 = logits[e]; i2 = e; }
  const float t  = __expf(m2 - m1);
  const float r  = __builtin_amdgcn_rcpf(1.f + t);
  if (lane == 0) {
    int p = atomicAdd(&cnt[i1], 1);
    list_slot[i1 * N_TOK + p] = 2 * n;
    slot_w[2 * n] = r;
    p = atomicAdd(&cnt[i2], 1);
    list_slot[i2 * N_TOK + p] = 2 * n + 1;
    slot_w[2 * n + 1] = t * r;
  }
}

// ------------- Kernel 2: per-expert gate/up GEMM (bf16 WMMA) + SiLU -------------
// Tile: M=32 tokens x 128 F-cols (gate) + 128 (up). 8 waves; each wave owns one
// 16-col subtile of gate and of up, two M-fragments (4 accums). Ping-pong LDS,
// register-prefetched global loads overlap the WMMA stage (single barrier/stage).
#define A_SZ (32 * 64)
#define B_SZ (256 * 64)
__global__ __launch_bounds__(256) void moe_gateup(const float* __restrict__ x,
                                                  const float* __restrict__ wgu,
                                                  const int* __restrict__ cnt,
                                                  const int* __restrict__ list_slot,
                                                  unsigned short* __restrict__ hbuf) {
  const int e = blockIdx.z;
  const int c = cnt[e];
  if ((int)blockIdx.y * 32 >= c) return;          // sparse: inactive tiles exit
  const int fb = blockIdx.x * 128;
  __shared__ __align__(16) unsigned short lA[2 * A_SZ];   // 8 KB
  __shared__ __align__(16) unsigned short lB[2 * B_SZ];   // 64 KB (gate 0..127, up 128..255)
  __shared__ int s_slot[32];
  __shared__ int s_tok[32];
  const int tid = threadIdx.x;
  if (tid < 32) {
    int p = blockIdx.y * 32 + tid;
    int slot = (p < c) ? list_slot[e * N_TOK + p] : -1;
    s_slot[tid] = slot;
    s_tok[tid] = (slot >= 0) ? (slot >> 1) : 0;
  }
  __syncthreads();
  const float* wbase = wgu + (size_t)e * TWOF * HDIM;
  const int wv = tid >> 5, lane = tid & 31;
  v8f aG0 = {0.f,0.f,0.f,0.f,0.f,0.f,0.f,0.f};
  v8f aG1 = aG0, aU0 = aG0, aU1 = aG0;

  // hoisted staging geometry
  const int ar = (tid * 8) >> 6, ac = (tid * 8) & 63;    // A: 8 floats/thread
  const float* asrc = x + (size_t)s_tok[ar] * HDIM + ac;
  const int brr = tid >> 2, bcc = (tid & 3) * 16;        // B: 16 floats/thread/pass
  const float* bsrc[4];
  int bdst[4];
#pragma unroll
  for (int pass = 0; pass < 4; ++pass) {
    int r  = pass * 64 + brr;
    int fg = (r < 128) ? (fb + r) : (FDIM + fb + (r - 128));
    bsrc[pass] = wbase + (size_t)fg * HDIM + bcc;
    bdst[pass] = r * 64 + bcc;
  }
  const int arow = lane & 15, acol = (lane >> 4) * 8;
  const int brow = lane & 15, bcol = (lane >> 4) * 8;

  float4 ga0, ga1, gb[16];
  auto load_stage = [&](int k0) {
    const float4* xs = (const float4*)(asrc + k0);
    ga0 = xs[0]; ga1 = xs[1];
#pragma unroll
    for (int pass = 0; pass < 4; ++pass) {
      const float4* src = (const float4*)(bsrc[pass] + k0);
#pragma unroll
      for (int q = 0; q < 4; ++q) gb[pass * 4 + q] = src[q];
    }
  };
  auto store_stage = [&](int p) {
    *(uint4*)&lA[p * A_SZ + ar * 64 + ac] = pk8bf(ga0, ga1);
#pragma unroll
    for (int pass = 0; pass < 4; ++pass) {
      *(uint4*)&lB[p * B_SZ + bdst[pass]]     = pk8bf(gb[pass*4+0], gb[pass*4+1]);
      *(uint4*)&lB[p * B_SZ + bdst[pass] + 8] = pk8bf(gb[pass*4+2], gb[pass*4+3]);
    }
  };

  load_stage(0);
  store_stage(0);
  int p = 0;
  for (int k0 = 0; k0 < HDIM; k0 += 64) {
    const bool has_next = (k0 + 64 < HDIM);
    __syncthreads();                       // buf[p] ready; buf[p^1] free
    if (has_next) load_stage(k0 + 64);     // global loads fly under the WMMAs
#pragma unroll
    for (int kk = 0; kk < 64; kk += 32) {
      const unsigned short* bA = &lA[p * A_SZ];
      const unsigned short* bB = &lB[p * B_SZ];
      v16bf a0 = lds_load_a(&bA[arow * 64 + kk + acol]);
      v16bf a1 = lds_load_a(&bA[(16 + arow) * 64 + kk + acol]);
      v16bf bg = lds_load_b(&bB[(wv * 16 + brow) * 64 + kk + bcol]);
      v16bf bu = lds_load_b(&bB[(128 + wv * 16 + brow) * 64 + kk + bcol]);
      aG0 = WMMA_BF16(a0, bg, aG0);
      aG1 = WMMA_BF16(a1, bg, aG1);
      aU0 = WMMA_BF16(a0, bu, aU0);
      aU1 = WMMA_BF16(a1, bu, aU1);
    }
    if (has_next) store_stage(p ^ 1);
    p ^= 1;
  }
  // epilogue: h = silu(gate)*up via fast rcp, store bf16 per routed slot
  const int ncol = lane & 15, mh = lane >> 4;
  const int col = fb + wv * 16 + ncol;
#pragma unroll
  for (int frag = 0; frag < 2; ++frag) {
    v8f g8 = frag ? aG1 : aG0;
    v8f u8 = frag ? aU1 : aU0;
#pragma unroll
    for (int r = 0; r < 8; ++r) {
      int m = frag * 16 + r + 8 * mh;
      int slot = s_slot[m];
      if (slot >= 0) {
        float g = g8[r], u = u8[r];
        float hv = g * u * __builtin_amdgcn_rcpf(1.f + __expf(-g));
        hbuf[(size_t)slot * FDIM + col] = f2bf(hv);
      }
    }
  }
}

// ------------- Kernel 3: per-expert down GEMM (bf16 WMMA), per-slot output -------------
// Tile: M=32 slots x 128 H-cols; same ping-pong pipeline.
#define DB_SZ (128 * 64)
__global__ __launch_bounds__(256) void moe_down(const unsigned short* __restrict__ hbuf,
                                                const float* __restrict__ wdn,
                                                const int* __restrict__ cnt,
                                                const int* __restrict__ list_slot,
                                                float* __restrict__ ypart) {
  const int e = blockIdx.z;
  const int c = cnt[e];
  if ((int)blockIdx.y * 32 >= c) return;
  const int jb = blockIdx.x * 128;
  __shared__ __align__(16) unsigned short lA[2 * A_SZ];    // 8 KB (bf16 h-tile)
  __shared__ __align__(16) unsigned short lB[2 * DB_SZ];   // 32 KB
  __shared__ int s_slot[32];
  const int tid = threadIdx.x;
  if (tid < 32) {
    int p = blockIdx.y * 32 + tid;
    s_slot[tid] = (p < c) ? list_slot[e * N_TOK + p] : -1;
  }
  __syncthreads();
  const float* wbase = wdn + (size_t)e * HDIM * FDIM;
  const int wv = tid >> 5, lane = tid & 31;
  v8f acc0 = {0.f,0.f,0.f,0.f,0.f,0.f,0.f,0.f};
  v8f acc1 = acc0;

  const int ar = (tid * 8) >> 6, ac = (tid * 8) & 63;   // A: 8 halfs/thread
  int aslot = s_slot[ar]; if (aslot < 0) aslot = 0;
  const unsigned short* asrc = hbuf + (size_t)aslot * FDIM + ac;
  const int brr = tid >> 2, bcc = (tid & 3) * 16;
  const float* bsrc[2];
  int bdst[2];
#pragma unroll
  for (int pass = 0; pass < 2; ++pass) {
    int r = pass * 64 + brr;
    bsrc[pass] = wbase + (size_t)(jb + r) * FDIM + bcc;
    bdst[pass] = r * 64 + bcc;
  }
  const int arow = lane & 15, acol = (lane >> 4) * 8;
  const int brow = lane & 15, bcol = (lane >> 4) * 8;

  uint4 gha;
  float4 gb[8];
  auto load_stage = [&](int k0) {
    gha = *(const uint4*)(asrc + k0);
#pragma unroll
    for (int pass = 0; pass < 2; ++pass) {
      const float4* src = (const float4*)(bsrc[pass] + k0);
#pragma unroll
      for (int q = 0; q < 4; ++q) gb[pass * 4 + q] = src[q];
    }
  };
  auto store_stage = [&](int p) {
    *(uint4*)&lA[p * A_SZ + ar * 64 + ac] = gha;
#pragma unroll
    for (int pass = 0; pass < 2; ++pass) {
      *(uint4*)&lB[p * DB_SZ + bdst[pass]]     = pk8bf(gb[pass*4+0], gb[pass*4+1]);
      *(uint4*)&lB[p * DB_SZ + bdst[pass] + 8] = pk8bf(gb[pass*4+2], gb[pass*4+3]);
    }
  };

  load_stage(0);
  store_stage(0);
  int p = 0;
  for (int k0 = 0; k0 < FDIM; k0 += 64) {
    const bool has_next = (k0 + 64 < FDIM);
    __syncthreads();
    if (has_next) load_stage(k0 + 64);
#pragma unroll
    for (int kk = 0; kk < 64; kk += 32) {
      const unsigned short* bA = &lA[p * A_SZ];
      const unsigned short* bB = &lB[p * DB_SZ];
      v16bf a0 = lds_load_a(&bA[arow * 64 + kk + acol]);
      v16bf a1 = lds_load_a(&bA[(16 + arow) * 64 + kk + acol]);
      v16bf b  = lds_load_b(&bB[(wv * 16 + brow) * 64 + kk + bcol]);
      acc0 = WMMA_BF16(a0, b, acc0);
      acc1 = WMMA_BF16(a1, b, acc1);
    }
    if (has_next) store_stage(p ^ 1);
    p ^= 1;
  }
  const int ncol = lane & 15, mh = lane >> 4;
  const int col = jb + wv * 16 + ncol;
#pragma unroll
  for (int frag = 0; frag < 2; ++frag) {
    v8f a8 = frag ? acc1 : acc0;
#pragma unroll
    for (int r = 0; r < 8; ++r) {
      int m = frag * 16 + r + 8 * mh;
      int slot = s_slot[m];
      if (slot >= 0) ypart[(size_t)slot * HDIM + col] = a8[r];
    }
  }
}

// ---------------- Kernel 4: combine the two expert contributions per token ----------------
__global__ __launch_bounds__(256) void moe_combine(const float* __restrict__ ypart,
                                                   const float* __restrict__ slot_w,
                                                   float* __restrict__ out) {
  const int idx4 = (blockIdx.x * 256 + threadIdx.x) * 4;
  const int n = idx4 >> 10;          // H = 1024
  const int j = idx4 & 1023;
  const float w0 = slot_w[2 * n];
  const float w1 = slot_w[2 * n + 1];
  const float4 a = *(const float4*)(ypart + (size_t)(2 * n)     * HDIM + j);
  const float4 b = *(const float4*)(ypart + (size_t)(2 * n + 1) * HDIM + j);
  float4 o;
  o.x = w0 * a.x + w1 * b.x;
  o.y = w0 * a.y + w1 * b.y;
  o.z = w0 * a.z + w1 * b.z;
  o.w = w0 * a.w + w1 * b.w;
  *(float4*)(out + idx4) = o;
}

extern "C" void kernel_launch(void* const* d_in, const int* in_sizes, int n_in,
                              void* d_out, int out_size, void* d_ws, size_t ws_size,
                              hipStream_t stream) {
  (void)in_sizes; (void)n_in; (void)out_size; (void)ws_size;
  const float* x   = (const float*)d_in[0];
  const float* gw  = (const float*)d_in[1];
  const float* wgu = (const float*)d_in[2];
  const float* wdn = (const float*)d_in[3];
  // d_in[4] = top_k (always 2 here)

  char* ws = (char*)d_ws;
  int*   cnt       = (int*)(ws + 0);                                  // 16 ints
  int*   list_slot = (int*)(ws + 256);                                // E*N ints (128 KB)
  float* slot_w    = (float*)(ws + 256 + (size_t)NEXP * N_TOK * 4);   // 2N floats
  unsigned short* hbuf = (unsigned short*)(ws + 147712);              // 2N x F bf16 (11.5 MB)
  float* ypart     = (float*)(ws + 147712 + (size_t)2 * N_TOK * FDIM * 2); // 2N x H f32 (16.8 MB)

  hipMemsetAsync(cnt, 0, 64, stream);
  moe_gate<<<N_TOK / 8, 256, 0, stream>>>(x, gw, cnt, list_slot, slot_w);
  moe_gateup<<<dim3(FDIM / 128, N_TOK / 32, NEXP), 256, 0, stream>>>(x, wgu, cnt, list_slot, hbuf);
  moe_down<<<dim3(HDIM / 128, N_TOK / 32, NEXP), 256, 0, stream>>>(hbuf, wdn, cnt, list_slot, ypart);
  moe_combine<<<(N_TOK * HDIM) / 1024, 256, 0, stream>>>(ypart, slot_w, (float*)d_out);
}
